// ImageEncoder_83923660964615
// MI455X (gfx1250) — compile-verified, matrix operand order checked by reading, not run on hardware
//
#include <hip/hip_runtime.h>
#include <hip/hip_bf16.h>

typedef __bf16 bf16;
typedef bf16  v16bf __attribute__((ext_vector_type(16)));
typedef float v8f   __attribute__((ext_vector_type(8)));
typedef unsigned int u32;
typedef u32 v4u __attribute__((ext_vector_type(4)));
typedef u32 v8u __attribute__((ext_vector_type(8)));

static inline int ceil_div(int a, int b) { return (a + b - 1) / b; }

__device__ __forceinline__ float siluf(float x) { return x / (1.f + expf(-x)); }

__device__ __forceinline__ v16bf pack_bf16(float4 x0, float4 x1, float4 x2, float4 x3)
{
    v16bf r;
    r[0]  = (bf16)x0.x; r[1]  = (bf16)x0.y; r[2]  = (bf16)x0.z; r[3]  = (bf16)x0.w;
    r[4]  = (bf16)x1.x; r[5]  = (bf16)x1.y; r[6]  = (bf16)x1.z; r[7]  = (bf16)x1.w;
    r[8]  = (bf16)x2.x; r[9]  = (bf16)x2.y; r[10] = (bf16)x2.z; r[11] = (bf16)x2.w;
    r[12] = (bf16)x3.x; r[13] = (bf16)x3.y; r[14] = (bf16)x3.z; r[15] = (bf16)x3.w;
    return r;
}

// ---------------------------------------------------------------------------
// FAST WMMA GEMM with Tensor-Data-Mover staged weights.
//   C[M,N] = A[M,K] @ W[N,K]^T (+bias) (+resid)
// Preconditions: M%64==0, N%32==0, K%KC==0, KC in {32,64,128} (template).
// Block = 4 waves, block tile 64(M) x 32(N); wave computes 16x32 (2 WMMAs per
// K-step, sharing the A fragment). The 32 W rows used by the whole block are
// DMA'd global->LDS by the TDM (one D# per K-chunk, double buffered,
// completion via TENSORcnt). D# pad feature inserts 4 dwords per KC-dword row
// -> LDS row stride KC+4 floats (bank-conflict-free column reads).
// KC is a template parameter so the inner K loop fully unrolls: up to 8
// back-to-back WMMAs per chunk with all LDS/global loads hoisted in front.
// ---------------------------------------------------------------------------
template <int KC>
__global__ __launch_bounds__(128)
void gemm_bf16_wmma_tdm(const float* __restrict__ A, int lda,
                        const float* __restrict__ W, int ldw,
                        float* __restrict__ C, int ldc,
                        const float* __restrict__ bias,
                        const float* __restrict__ resid,
                        int M, int N, int K)
{
    constexpr int RSTRIDE  = KC + 4;
    constexpr u32 PAD_CODE = (KC == 32) ? 4u : (KC == 64) ? 5u : 6u; // interval=KC dwords
    __shared__ float sW[2][32 * RSTRIDE];

    const int lane = threadIdx.x & 31;
    const int wv   = threadIdx.x >> 5;
    const int nb_n = N >> 5;
    const int bn   = blockIdx.x % nb_n;
    const int bm   = blockIdx.x / nb_n;
    const int tm   = bm * 4 + wv;              // this wave's 16-row tile
    const int mrow = tm * 16 + (lane & 15);
    const int kb   = (lane >> 4) << 3;         // per-lane K chunk base (0 or 8)
    const int nchunks = K / KC;

    const float* Wbase = W + (size_t)(bn << 5) * ldw;

    auto issue_chunk = [&](int c, int buf) {
        // Tensor DMA Descriptor (D#), 2D tile: KC x 32 rows of W.
        unsigned long long ga = (unsigned long long)(Wbase + (size_t)c * KC);
        u32 lds = (u32)(unsigned long long)(&sW[buf][0]);
        u32 rem = (u32)(K - c * KC);           // tensor_dim0 (in-bounds bound)
        v4u g0; v8u g1; v4u g2; v4u g3;
        g0[0] = 1u;                                           // count=1
        g0[1] = lds;                                          // lds_addr (bytes)
        g0[2] = (u32)ga;                                      // global_addr lo
        g0[3] = ((u32)(ga >> 32) & 0x01FFFFFFu) | (2u << 30); // addr hi | type=2
        g1[0] = (2u << 16)                  // data_size = 4B
              | (1u << 20)                  // pad_enable
              | (PAD_CODE << 22)            // pad_interval = KC dwords
              | (3u << 25);                 // pad_amount = 4 dwords
        g1[1] = (rem & 0xffffu) << 16;                        // tensor_dim0 lo
        g1[2] = (rem >> 16) | (32u << 16);                    // dim0 hi | dim1=32
        g1[3] = ((u32)KC << 16);                              // tile_dim0 = KC
        g1[4] = 32u;                                          // tile_dim1=32, tile_dim2=0
        g1[5] = (u32)ldw;                                     // dim0_stride lo
        g1[6] = 0u;
        g1[7] = 0u;
        g2[0] = 1u; g2[1] = 0u; g2[2] = 0u; g2[3] = 0u;       // tensor_dim2=1
        g3[0] = 0u; g3[1] = 0u; g3[2] = 0u; g3[3] = 0u;
        asm volatile("tensor_load_to_lds %0, %1, %2, %3"
                     :: "s"(g0), "s"(g1), "s"(g2), "s"(g3) : "memory");
    };

    if (wv == 0) issue_chunk(0, 0);

    v8f acc0 = {0.f,0.f,0.f,0.f,0.f,0.f,0.f,0.f};
    v8f acc1 = {0.f,0.f,0.f,0.f,0.f,0.f,0.f,0.f};
    const float* Arow = A + (size_t)mrow * lda;

    for (int c = 0; c < nchunks; ++c) {
        if (wv == 0) {
            if (c + 1 < nchunks) {
                issue_chunk(c + 1, (c + 1) & 1);
                __builtin_amdgcn_s_wait_tensorcnt(1);  // chunk c complete
            } else {
                __builtin_amdgcn_s_wait_tensorcnt(0);
            }
        }
        __syncthreads();

        const float* Wb = &sW[c & 1][0];
        const int kg = c * KC;
#pragma unroll
        for (int k0 = 0; k0 < KC; k0 += 32) {
            // A fragment: 16 floats, 4x b128 from global
            const float* ap = Arow + kg + k0 + kb;
            float4 a0 = *(const float4*)(ap);
            float4 a1 = *(const float4*)(ap + 4);
            float4 a2 = *(const float4*)(ap + 16);
            float4 a3 = *(const float4*)(ap + 20);
            // W fragments for two 16-col tiles: 4x b128 each from LDS
            const float* w0 = Wb + (lane & 15) * RSTRIDE + k0 + kb;
            const float* w1 = w0 + 16 * RSTRIDE;
            float4 b0 = *(const float4*)(w0);
            float4 b1 = *(const float4*)(w0 + 4);
            float4 b2 = *(const float4*)(w0 + 16);
            float4 b3 = *(const float4*)(w0 + 20);
            float4 c0 = *(const float4*)(w1);
            float4 c1 = *(const float4*)(w1 + 4);
            float4 c2 = *(const float4*)(w1 + 16);
            float4 c3 = *(const float4*)(w1 + 20);
            v16bf av = pack_bf16(a0, a1, a2, a3);
            v16bf bv = pack_bf16(b0, b1, b2, b3);
            v16bf cv = pack_bf16(c0, c1, c2, c3);
            acc0 = __builtin_amdgcn_wmma_f32_16x16x32_bf16(
                       false, av, false, bv, (short)0, acc0, false, false);
            acc1 = __builtin_amdgcn_wmma_f32_16x16x32_bf16(
                       false, av, false, cv, (short)0, acc1, false, false);
        }
        __syncthreads();
    }

    const int n0 = (bn << 5) + (lane & 15);
    const int mb = tm * 16 + ((lane >> 4) << 3);
#pragma unroll
    for (int v = 0; v < 8; ++v) {
        float v0 = acc0[v], v1 = acc1[v];
        if (bias)  { v0 += bias[n0];  v1 += bias[n0 + 16]; }
        const size_t row = (size_t)(mb + v) * ldc;
        if (resid) { v0 += resid[row + n0]; v1 += resid[row + n0 + 16]; }
        C[row + n0]      = v0;
        C[row + n0 + 16] = v1;
    }
}

// ---------------------------------------------------------------------------
// EDGE WMMA GEMM (arbitrary M/N/K, zero-padded, optional softplus).
// Loads are unconditional from clamped addresses and multiplied by a 0/1
// mask so the compiler does not emit exec-masked branchy loads.
// ---------------------------------------------------------------------------
template <int ACT> // 0=none, 1=softplus
__global__ __launch_bounds__(128)
void gemm_bf16_wmma_edge(const float* __restrict__ A, int lda,
                         const float* __restrict__ W, int ldw,
                         float* __restrict__ C, int ldc,
                         const float* __restrict__ bias,
                         const float* __restrict__ resid,
                         int M, int N, int K)
{
    const int lane = threadIdx.x & 31;
    const int wv   = threadIdx.x >> 5;
    const int tiles_n = (N + 15) >> 4;
    const int tiles_m = (M + 15) >> 4;
    const int tile = blockIdx.x * 4 + wv;
    if (tile >= tiles_m * tiles_n) return;
    const int tm = tile / tiles_n;
    const int tn = tile - tm * tiles_n;

    const int mrow = tm * 16 + (lane & 15);
    const int ncol = tn * 16 + (lane & 15);
    const bool mok = mrow < M;
    const bool nok = ncol < N;
    const float* Ar = A + (size_t)(mok ? mrow : 0) * lda;
    const float* Wr = W + (size_t)(nok ? ncol : 0) * ldw;
    const int kb = (lane >> 4) << 3;

    v8f acc = {0.f,0.f,0.f,0.f,0.f,0.f,0.f,0.f};

    for (int k0 = 0; k0 < K; k0 += 32) {
        v16bf a, b;
#pragma unroll
        for (int i = 0; i < 8; ++i) {
            const int k1 = k0 + kb + i;
            const int k2 = k1 + 16;
            const int k1c = k1 < K ? k1 : 0;
            const int k2c = k2 < K ? k2 : 0;
            const float am1 = (float)(int)(mok & (k1 < K));
            const float am2 = (float)(int)(mok & (k2 < K));
            const float wm1 = (float)(int)(nok & (k1 < K));
            const float wm2 = (float)(int)(nok & (k2 < K));
            a[i]     = (bf16)(Ar[k1c] * am1);
            a[i + 8] = (bf16)(Ar[k2c] * am2);
            b[i]     = (bf16)(Wr[k1c] * wm1);
            b[i + 8] = (bf16)(Wr[k2c] * wm2);
        }
        if (k0 + 32 < K) {
            __builtin_prefetch(Ar + k0 + 32 + kb, 0, 0);  // global_prefetch_b8
            __builtin_prefetch(Wr + k0 + 32 + kb, 0, 0);
        }
        acc = __builtin_amdgcn_wmma_f32_16x16x32_bf16(
                  false, a, false, b, (short)0, acc, false, false);
    }

    if (!nok) return;
    const int mbase = tm * 16 + ((lane >> 4) << 3);
#pragma unroll
    for (int v = 0; v < 8; ++v) {
        const int cm = mbase + v;
        if (cm < M) {
            float val = acc[v];
            if (bias)    val += bias[ncol];
            if (ACT == 1) val = (val > 20.f) ? val : log1pf(expf(val));
            if (resid)   val += resid[(size_t)cm * ldc + ncol];
            C[(size_t)cm * ldc + ncol] = val;
        }
    }
}

// ---------------------------------------------------------------------------
// LayerNorm over last dim: one wave32 per token, shuffle reductions.
// ---------------------------------------------------------------------------
__global__ __launch_bounds__(256)
void layernorm_k(const float* __restrict__ x, float* __restrict__ out,
                 const float* __restrict__ g, const float* __restrict__ b,
                 int L, int d)
{
    const int tok = blockIdx.x * (blockDim.x >> 5) + (threadIdx.x >> 5);
    if (tok >= L) return;
    const int lane = threadIdx.x & 31;
    const float* xr = x + (size_t)tok * d;
    float s = 0.f, s2 = 0.f;
    for (int c = lane; c < d; c += 32) { float v = xr[c]; s += v; s2 += v * v; }
#pragma unroll
    for (int o = 16; o > 0; o >>= 1) { s += __shfl_down(s, o); s2 += __shfl_down(s2, o); }
    s = __shfl(s, 0); s2 = __shfl(s2, 0);
    const float mean = s / d;
    const float var  = s2 / d - mean * mean;
    const float inv  = rsqrtf(var + 1e-5f);
    float* orow = out + (size_t)tok * d;
    for (int c = lane; c < d; c += 32)
        orow[c] = (xr[c] - mean) * inv * g[c] + b[c];
}

// Patch embed: 4x4/s4 conv, 3->96ch, 256x256 -> (64*64, 96).
__global__ __launch_bounds__(256)
void patch_embed_k(const float* __restrict__ x, const float* __restrict__ w,
                   const float* __restrict__ b, float* __restrict__ out)
{
    const int idx = blockIdx.x * blockDim.x + threadIdx.x;
    if (idx >= 64 * 64 * 96) return;
    const int c = idx % 96;
    const int p = idx / 96;
    const int i = p >> 6, j = p & 63;
    float acc = b[c];
    const float* wc = w + c * 48;
#pragma unroll
    for (int ic = 0; ic < 3; ++ic)
#pragma unroll
        for (int u = 0; u < 4; ++u)
#pragma unroll
            for (int v = 0; v < 4; ++v)
                acc += wc[(ic * 4 + u) * 4 + v] *
                       x[(ic * 256 + i * 4 + u) * 256 + j * 4 + v];
    out[idx] = acc;
}

// Depthwise causal conv1d (k=3) + bias + SiLU.
__global__ __launch_bounds__(256)
void dwconv_silu_k(const float* __restrict__ xz, int ldxz,
                   const float* __restrict__ cw, const float* __restrict__ cb,
                   float* __restrict__ xc, int L, int di)
{
    const int idx = blockIdx.x * blockDim.x + threadIdx.x;
    if (idx >= L * di) return;
    const int c = idx % di, t = idx / di;
    float acc = cb[c];
#pragma unroll
    for (int j = 0; j < 3; ++j) {
        const int tt = t - 2 + j;
        if (tt >= 0) acc += cw[c * 3 + j] * xz[(size_t)tt * ldxz + c];
    }
    xc[idx] = siluf(acc);
}

// Fused selective scan: one thread per channel, 16 states in registers.
__global__ __launch_bounds__(256)
void selective_scan_k(const float* __restrict__ dt,
                      const float* __restrict__ xc,
                      const float* __restrict__ xd,
                      int ldxd, int dtr,
                      const float* __restrict__ A_log,
                      const float* __restrict__ Dp,
                      const float* __restrict__ xz, int ldxz,
                      float* __restrict__ y, int L, int di)
{
    const int c = blockIdx.x * blockDim.x + threadIdx.x;
    if (c >= di) return;
    float A[16], h[16];
#pragma unroll
    for (int n = 0; n < 16; ++n) { A[n] = -expf(A_log[c * 16 + n]); h[n] = 0.f; }
    const float Dc = Dp[c];
    for (int t = 0; t < L; ++t) {
        const float dtv = dt[(size_t)t * di + c];
        const float xv  = xc[(size_t)t * di + c];
        const float* Bv = xd + (size_t)t * ldxd + dtr;
        const float* Cv = Bv + 16;
        const float dx = dtv * xv;
        float acc = 0.f;
#pragma unroll
        for (int n = 0; n < 16; ++n) {
            h[n] = expf(dtv * A[n]) * h[n] + dx * Bv[n];
            acc += h[n] * Cv[n];
        }
        const float z = xz[(size_t)t * ldxz + di + c];
        y[(size_t)t * di + c] = (acc + Dc * xv) * siluf(z);
    }
}

// 4-direction permutation gather (input side).
__global__ __launch_bounds__(256)
void gather_dir_k(const float* __restrict__ h, float* __restrict__ o,
                  int hw, int d, int dir)
{
    const int L = hw * hw;
    const int idx = blockIdx.x * blockDim.x + threadIdx.x;
    if (idx >= L * d) return;
    const int c = idx % d, p = idx / d;
    const int i = p / hw, j = p % hw;
    int src;
    if      (dir == 0) src = p;
    else if (dir == 1) src = (hw - 1 - j) * hw + i;
    else if (dir == 2) src = L - 1 - p;
    else { const int q = L - 1 - p; const int qi = q / hw, qj = q % hw;
           src = (hw - 1 - qj) * hw + qi; }
    o[idx] = h[(size_t)src * d + c];
}

// Combine the four direction outputs with the inverse permutations.
__global__ __launch_bounds__(256)
void combine4_k(const float* __restrict__ m0, const float* __restrict__ m1,
                const float* __restrict__ m2, const float* __restrict__ m3,
                float* __restrict__ o, int hw, int d)
{
    const int L = hw * hw;
    const int idx = blockIdx.x * blockDim.x + threadIdx.x;
    if (idx >= L * d) return;
    const int c = idx % d, t = idx / d;
    const int i = t / hw, j = t % hw;
    const int p1 = j * hw + (hw - 1 - i);
    const int p2 = L - 1 - t;
    const int p3 = L - 1 - p1;
    o[idx] = m0[(size_t)t  * d + c] + m1[(size_t)p1 * d + c] +
             m2[(size_t)p2 * d + c] + m3[(size_t)p3 * d + c];
}

// Patch merge 2x2 -> concat [even/even, odd/even, even/odd, odd/odd].
__global__ __launch_bounds__(256)
void merge_gather_k(const float* __restrict__ h, float* __restrict__ o,
                    int hw, int d)
{
    const int hw2 = hw >> 1, L2 = hw2 * hw2, d4 = d << 2;
    const int idx = blockIdx.x * blockDim.x + threadIdx.x;
    if (idx >= L2 * d4) return;
    const int q = idx % d4, p = idx / d4;
    const int grp = q / d, c = q % d;
    const int i = p / hw2, j = p % hw2;
    const int r  = 2 * i + ((grp == 1) || (grp == 3));
    const int cc = 2 * j + ((grp == 2) || (grp == 3));
    o[idx] = h[((size_t)(r * hw + cc)) * d + c];
}

// (L, d) token map -> (d, hw, hw) channel-major feature map.
__global__ __launch_bounds__(256)
void extract_chw_k(const float* __restrict__ h, float* __restrict__ o,
                   int hw, int d)
{
    const int L = hw * hw;
    const int idx = blockIdx.x * blockDim.x + threadIdx.x;
    if (idx >= L * d) return;
    const int c = idx % d, p = idx / d;
    o[(size_t)c * L + p] = h[(size_t)p * d + c];
}

// Classifier head.
__global__ void cls_head_k(const float* __restrict__ lnx, const float* __restrict__ w,
                           const float* __restrict__ bias, float* __restrict__ out)
{
    const int lane = threadIdx.x & 31;
    const int o    = threadIdx.x >> 5;
    if (o >= 6) return;
    float s = 0.f;
    for (int c = lane; c < 768; c += 32) {
        float m = 0.f;
        for (int t = 0; t < 64; ++t) m += lnx[t * 768 + c];
        s += (m * (1.f / 64.f)) * w[o * 768 + c];
    }
#pragma unroll
    for (int off = 16; off > 0; off >>= 1) s += __shfl_down(s, off);
    if (lane == 0) out[o] = s + bias[o];
}

// ---------------------------------------------------------------------------
// Host-side orchestration
// ---------------------------------------------------------------------------
struct BlockP {
    const float *in_w, *conv_w, *conv_b, *xproj_w, *dt_w, *dt_b,
                *A_log, *D, *out_w, *ln_g, *ln_b, *proj_w, *proj_b;
};
struct MambaWs { float *xz, *xc, *xd, *dt, *ybuf; };

static void gemm(const float* A, int lda, const float* W, int ldw,
                 float* C, int ldc, const float* bias, const float* resid,
                 int M, int N, int K, int act, hipStream_t s)
{
    if (act == 0 && (M % 64 == 0) && (N % 32 == 0) && (K % 32 == 0)) {
        const int nb = (M / 64) * (N / 32);
        if      (K % 128 == 0) gemm_bf16_wmma_tdm<128><<<nb, 128, 0, s>>>(A, lda, W, ldw, C, ldc, bias, resid, M, N, K);
        else if (K % 64  == 0) gemm_bf16_wmma_tdm<64 ><<<nb, 128, 0, s>>>(A, lda, W, ldw, C, ldc, bias, resid, M, N, K);
        else                   gemm_bf16_wmma_tdm<32 ><<<nb, 128, 0, s>>>(A, lda, W, ldw, C, ldc, bias, resid, M, N, K);
        return;
    }
    const int tiles = ((M + 15) / 16) * ((N + 15) / 16);
    const int nb = (tiles + 3) / 4;
    if (act) gemm_bf16_wmma_edge<1><<<nb, 128, 0, s>>>(A, lda, W, ldw, C, ldc, bias, resid, M, N, K);
    else     gemm_bf16_wmma_edge<0><<<nb, 128, 0, s>>>(A, lda, W, ldw, C, ldc, bias, resid, M, N, K);
}

static void run_mamba(const BlockP& bp, const float* xin, float* mout,
                      int L, int d, const MambaWs& w, hipStream_t s)
{
    const int di   = 2 * d;
    const int dtr  = (d + 15) / 16;
    const int ldxd = dtr + 32;
    gemm(xin, d, bp.in_w, d, w.xz, 2 * di, nullptr, nullptr, L, 2 * di, d, 0, s);
    dwconv_silu_k<<<ceil_div(L * di, 256), 256, 0, s>>>(w.xz, 2 * di, bp.conv_w, bp.conv_b, w.xc, L, di);
    gemm(w.xc, di, bp.xproj_w, di, w.xd, ldxd, nullptr, nullptr, L, ldxd, di, 0, s);
    gemm(w.xd, ldxd, bp.dt_w, dtr, w.dt, di, bp.dt_b, nullptr, L, di, dtr, 1, s);
    selective_scan_k<<<ceil_div(di, 256), 256, 0, s>>>(
        w.dt, w.xc, w.xd, ldxd, dtr, bp.A_log, bp.D, w.xz, 2 * di, w.ybuf, L, di);
    gemm(w.ybuf, di, bp.out_w, di, mout, d, nullptr, nullptr, L, d, di, 0, s);
}

static void run_block(const BlockP& bp, const float* h, float* hout,
                      int hw, int d, float* dirin, float** m, float* ysum,
                      float* lnbuf, const MambaWs& w, hipStream_t s)
{
    const int L = hw * hw;
    for (int k = 0; k < 4; ++k) {
        gather_dir_k<<<ceil_div(L * d, 256), 256, 0, s>>>(h, dirin, hw, d, k);
        run_mamba(bp, dirin, m[k], L, d, w, s);
    }
    combine4_k<<<ceil_div(L * d, 256), 256, 0, s>>>(m[0], m[1], m[2], m[3], ysum, hw, d);
    layernorm_k<<<ceil_div(L, 8), 256, 0, s>>>(ysum, lnbuf, bp.ln_g, bp.ln_b, L, d);
    gemm(lnbuf, d, bp.proj_w, d, hout, d, bp.proj_b, h, L, d, d, 0, s);
}

extern "C" void kernel_launch(void* const* d_in, const int* in_sizes, int n_in,
                              void* d_out, int out_size, void* d_ws, size_t ws_size,
                              hipStream_t stream)
{
    static const int DEPTH[4] = {2, 2, 5, 2};
    if (n_in < 175) return;

    // --- decode parameter pointers (setup_inputs() dict insertion order) ---
    const float** P = (const float**)alloca(sizeof(float*) * n_in);
    for (int i = 0; i < n_in; ++i) P[i] = (const float*)d_in[i];
    int ix = 0;
    const float* ximg  = P[ix++];
    const float* pe_w  = P[ix++]; const float* pe_b  = P[ix++];
    const float* pe_g  = P[ix++]; const float* pe_b2 = P[ix++];
    BlockP blocks[4][5];
    const float* ng[4][5]; const float* nb_[4][5];
    const float* mg[3]; const float* mb[3]; const float* mw[3];
    for (int i = 0; i < 4; ++i) {
        for (int j = 0; j < DEPTH[i]; ++j) {
            BlockP& b = blocks[i][j];
            b.in_w = P[ix++]; b.conv_w = P[ix++]; b.conv_b = P[ix++];
            b.xproj_w = P[ix++]; b.dt_w = P[ix++]; b.dt_b = P[ix++];
            b.A_log = P[ix++]; b.D = P[ix++]; b.out_w = P[ix++];
            b.ln_g = P[ix++]; b.ln_b = P[ix++]; b.proj_w = P[ix++]; b.proj_b = P[ix++];
        }
        for (int j = 0; j < DEPTH[i] - 1; ++j) { ng[i][j] = P[ix++]; nb_[i][j] = P[ix++]; }
        if (i < 3) { mg[i] = P[ix++]; mb[i] = P[ix++]; mw[i] = P[ix++]; }
    }
    const float* cls_g = P[ix++]; const float* cls_b = P[ix++];
    const float* cls_w = P[ix++]; const float* cls_bias = P[ix++];

    // --- workspace layout (floats) ---
    const size_t SZH = (size_t)4096 * 96;
    if (ws_size < 20 * SZH * sizeof(float)) return;
    float* wsf  = (float*)d_ws;
    float* bufA  = wsf;
    float* bufB  = wsf +  1 * SZH;
    float* dirin = wsf +  2 * SZH;
    float* ysum  = wsf +  3 * SZH;
    float* lnbuf = wsf +  4 * SZH;
    float* m[4]  = { wsf + 5 * SZH, wsf + 6 * SZH, wsf + 7 * SZH, wsf + 8 * SZH };
    MambaWs w;
    w.xz   = wsf +  9 * SZH;
    w.xc   = wsf + 13 * SZH;
    w.dt   = wsf + 15 * SZH;
    w.ybuf = wsf + 17 * SZH;
    w.xd   = wsf + 19 * SZH;

    float* dout = (float*)d_out;
    const size_t OFF_FINAL = 6;
    const size_t OFF_E0 = OFF_FINAL + (size_t)768 * 64;
    const size_t OFF_E1 = OFF_E0 + (size_t)96 * 64 * 64;
    const size_t OFF_E2 = OFF_E1 + (size_t)192 * 32 * 32;
    const size_t EOFF[3] = { OFF_E0, OFF_E1, OFF_E2 };

    hipStream_t s = stream;
    float* hc = bufA; float* hn = bufB;

    patch_embed_k<<<ceil_div(64 * 64 * 96, 256), 256, 0, s>>>(ximg, pe_w, pe_b, hc);
    layernorm_k<<<ceil_div(4096, 8), 256, 0, s>>>(hc, hn, pe_g, pe_b2, 4096, 96);
    { float* t = hc; hc = hn; hn = t; }

    int d = 96, hw = 64;
    for (int i = 0; i < 4; ++i) {
        const int L = hw * hw;
        for (int j = 0; j < DEPTH[i]; ++j) {
            run_block(blocks[i][j], hc, hn, hw, d, dirin, m, ysum, lnbuf, w, s);
            { float* t = hc; hc = hn; hn = t; }
            if (j < DEPTH[i] - 1) {
                layernorm_k<<<ceil_div(L, 8), 256, 0, s>>>(hc, hn, ng[i][j], nb_[i][j], L, d);
                { float* t = hc; hc = hn; hn = t; }
            }
        }
        if (i < 3) {
            extract_chw_k<<<ceil_div(L * d, 256), 256, 0, s>>>(hc, dout + EOFF[i], hw, d);
            const int hw2 = hw >> 1, L2 = hw2 * hw2, d4 = 4 * d;
            merge_gather_k<<<ceil_div(L2 * d4, 256), 256, 0, s>>>(hc, m[0], hw, d);
            layernorm_k<<<ceil_div(L2, 8), 256, 0, s>>>(m[0], lnbuf, mg[i], mb[i], L2, d4);
            gemm(lnbuf, d4, mw[i], d4, hn, 2 * d, nullptr, nullptr, L2, 2 * d, d4, 0, s);
            { float* t = hc; hc = hn; hn = t; }
            d *= 2; hw = hw2;
        }
    }

    extract_chw_k<<<ceil_div(64 * 768, 256), 256, 0, s>>>(hc, dout + OFF_FINAL, 8, 768);
    layernorm_k<<<ceil_div(64, 8), 256, 0, s>>>(hc, lnbuf, cls_g, cls_b, 64, 768);
    cls_head_k<<<1, 192, 0, s>>>(lnbuf, cls_w, cls_bias, dout);
}
